// SeqAttention_5128190951723
// MI455X (gfx1250) — matrix operand order repaired and checked by reading, not compiled, not run on hardware
//
#include <hip/hip_runtime.h>

// ---------------------------------------------------------------------------
// SeqAttention collapse:
//   alpha = softmax(e, axis=1); s = sum(alpha, axis=1)  ==>  s == 1 exactly.
//   context[b,t] = sum_j hid[b,t,j] = x[b,t,:] . colsum(W) + sum(b)
// So the whole pipeline is one memory-bound GEMV (256 MB of x @ 23.3 TB/s
// ~= 11 us floor). The dot products are routed through v_wmma_f32_16x16x32_f16
// (free: kernel is bandwidth-bound at 0.5 flop/byte), f32 accumulation.
// ---------------------------------------------------------------------------

typedef __attribute__((ext_vector_type(16))) _Float16 v16h;
typedef __attribute__((ext_vector_type(8)))  float    v8f;
typedef __attribute__((ext_vector_type(4)))  float    v4f;

#define B_DIM 32
#define T_DIM 2048
#define D_DIM 1024

// ---------------------------------------------------------------------------
// Kernel 1: wsum[d] = sum_e W[e][d]  (f32 accumulate, stored f16 for WMMA B),
//           bsum    = sum_e b[e].
// W is [D,D] row-major; threads over d -> fully coalesced column sums. 4 MB
// read, negligible runtime.
// ---------------------------------------------------------------------------
__global__ __launch_bounds__(256)
void colsum_kernel(const float* __restrict__ W, const float* __restrict__ bias,
                   _Float16* __restrict__ wsumh, float* __restrict__ bsum) {
  const int d = blockIdx.x * blockDim.x + threadIdx.x;
  if (d < D_DIM) {
    float acc = 0.0f;
#pragma unroll 8
    for (int e = 0; e < D_DIM; ++e) acc += W[(size_t)e * D_DIM + d];
    wsumh[d] = (_Float16)acc;
  }
  if (blockIdx.x == 0 && threadIdx.x == 0) {
    float s = 0.0f;
    for (int e = 0; e < D_DIM; ++e) s += bias[e];
    *bsum = s;
  }
}

// ---------------------------------------------------------------------------
// A-matrix (16x32 f16) lane layout (contiguous K order, matches B):
//   lane L (M = L%16): elements 0..15 -> K = klo+0..15,  klo = (L>=16)*16
// ---------------------------------------------------------------------------
__device__ __forceinline__ v16h load_a_tile(const float* __restrict__ xrow,
                                            int kbase, int klo) {
  const v4f* p = (const v4f*)(xrow + kbase + klo);
  const v4f* q = (const v4f*)(xrow + kbase + klo + 8);
  v4f a0 = __builtin_nontemporal_load(p);
  v4f a1 = __builtin_nontemporal_load(p + 1);
  v4f a2 = __builtin_nontemporal_load(q);
  v4f a3 = __builtin_nontemporal_load(q + 1);
  v16h a;
  a[0]  = (_Float16)a0[0]; a[1]  = (_Float16)a0[1];
  a[2]  = (_Float16)a0[2]; a[3]  = (_Float16)a0[3];
  a[4]  = (_Float16)a1[0]; a[5]  = (_Float16)a1[1];
  a[6]  = (_Float16)a1[2]; a[7]  = (_Float16)a1[3];
  a[8]  = (_Float16)a2[0]; a[9]  = (_Float16)a2[1];
  a[10] = (_Float16)a2[2]; a[11] = (_Float16)a2[3];
  a[12] = (_Float16)a3[0]; a[13] = (_Float16)a3[1];
  a[14] = (_Float16)a3[2]; a[15] = (_Float16)a3[3];
  return a;
}

// ---------------------------------------------------------------------------
// Kernel 2: out[r] = x[r,:] . wsum + bsum  for r in [0, B*T), via WMMA.
// One wave handles 16 consecutive rows. B-matrix = wsum chunk broadcast
// across all 16 columns (B layout: lanes 0-15 hold K 0..15, lanes 16-31 hold
// K 16..31 -> per-lane v16h is just 16 contiguous halves). All D columns of
// the WMMA result are identical; C/D layout puts rows 0-7 in lanes 0-15 and
// rows 8-15 in lanes 16-31, so lanes 0 and 16 emit the 16 outputs.
// ---------------------------------------------------------------------------
__global__ __launch_bounds__(256)
void seqattn_gemv_wmma(const float* __restrict__ x,
                       const _Float16* __restrict__ wsumh,
                       const float* __restrict__ bsum,
                       float* __restrict__ out) {
  const int lane = threadIdx.x & 31;
  const int wave = (blockIdx.x * blockDim.x + threadIdx.x) >> 5;
  const int r0   = wave * 16;                 // first of 16 flattened rows
  const int M    = lane & 15;
  const int hi   = lane >> 4;                 // 0 = lanes 0-15, 1 = lanes 16-31
  const int klo  = hi * 16;                   // A-layout K base for this half

  const float* xrow = x + (size_t)(r0 + M) * D_DIM;

  v8f acc0 = {};
  v8f acc1 = {};
  // Two accumulators: breaks the WMMA->WMMA RAW chain and doubles the number
  // of loads in flight (we are HBM-latency/bandwidth bound).
  for (int k0 = 0; k0 < D_DIM; k0 += 64) {
    v16h a0 = load_a_tile(xrow, k0, klo);
    v16h a1 = load_a_tile(xrow, k0 + 32, klo);
    v16h b0 = *(const v16h*)(wsumh + k0 + hi * 16);
    v16h b1 = *(const v16h*)(wsumh + k0 + 32 + hi * 16);
    acc0 = __builtin_amdgcn_wmma_f32_16x16x32_f16(
        false, a0, false, b0, (short)0, acc0, false, false);
    acc1 = __builtin_amdgcn_wmma_f32_16x16x32_f16(
        false, a1, false, b1, (short)0, acc1, false, false);
  }

  const float bs = *bsum;
  if (M == 0) {                               // lanes 0 (rows 0-7) and 16 (rows 8-15)
    float* o = out + r0 + hi * 8;
#pragma unroll
    for (int i = 0; i < 8; ++i) o[i] = acc0[i] + acc1[i] + bs;
  }
}

extern "C" void kernel_launch(void* const* d_in, const int* in_sizes, int n_in,
                              void* d_out, int out_size, void* d_ws, size_t ws_size,
                              hipStream_t stream) {
  (void)in_sizes; (void)n_in; (void)out_size; (void)ws_size;
  const float* x = (const float*)d_in[0];   // [B,T,D] f32
  const float* W = (const float*)d_in[1];   // [D,D]  f32
  const float* b = (const float*)d_in[2];   // [D]    f32
  float* out = (float*)d_out;               // [B,T]  f32

  _Float16* wsumh = (_Float16*)d_ws;                              // 2 KB
  float*    bsum  = (float*)((char*)d_ws + D_DIM * sizeof(_Float16));

  colsum_kernel<<<D_DIM / 256, 256, 0, stream>>>(W, b, wsumh, bsum);

  // B*T/16 = 4096 waves; 8 waves (256 threads) per block -> 512 blocks.
  const int nblocks = (B_DIM * T_DIM) / 16 / 8;
  seqattn_gemv_wmma<<<nblocks, 256, 0, stream>>>(x, wsumh, bsum, out);
}
